// Fusion_32693291057756
// MI455X (gfx1250) — matrix operand order repaired and checked by reading, not compiled, not run on hardware
//
#include <hip/hip_runtime.h>

// Problem constants (from reference)
#define N_K   1024
#define N_E   20480
#define D_IN  4096
#define D_OUT 1024
#define E_K   65536
#define E_B   131072
#define NB    (N_E + N_K)   // 21504

typedef __attribute__((ext_vector_type(8)))  __bf16 v8bf;
typedef __attribute__((ext_vector_type(16))) __bf16 v16bf;
typedef __attribute__((ext_vector_type(8)))  float  v8f;
typedef __attribute__((ext_vector_type(4)))  unsigned int u32x4;
typedef __attribute__((ext_vector_type(8)))  int i32x8;
typedef __attribute__((ext_vector_type(4)))  int i32x4;

// ------------------------- small utility kernels -------------------------

__global__ void cvt_f32_bf16(const float* __restrict__ in, __bf16* __restrict__ out, size_t n) {
    size_t i = (size_t)blockIdx.x * blockDim.x + threadIdx.x;
    size_t stride = (size_t)gridDim.x * blockDim.x;
    for (; i < n; i += stride) out[i] = (__bf16)in[i];
}

__global__ void fill_f32(float* __restrict__ p, float v, size_t n) {
    size_t i = (size_t)blockIdx.x * blockDim.x + threadIdx.x;
    size_t stride = (size_t)gridDim.x * blockDim.x;
    for (; i < n; i += stride) p[i] = v;
}

// ------------------------- WMMA bf16 GEMM: C[M,N] = A[M,K] * B[N,K]^T ----
// A row-major (M x K) bf16 streamed per-wave from global (no intra-block
// redundancy).  B tile (64 cols x 64 K) staged into LDS by the Tensor Data
// Mover, double buffered, overlapping DMA with WMMA.  Block = 8 waves ->
// 128(M) x 64(N) C tile; each wave owns a 16-row strip with 4 accumulators.

#define KSTAGE 64
#define BCOLS  64
#define BPITCH 72   // 64 data + 8 pad bf16 per row (144B) -> conflict-free b128 LDS reads

// TDM descriptor group-1: uniform for all stages of a given GEMM.
//  dw0: data_size=2B (code 1), pad_enable, pad_interval=32 DWORDs (code 4),
//       pad_amount=4 DWORDs (code 3)  -> hardware inserts 16B pad per 128B row
//  tensor_dim0 = K, tensor_dim1 = 64, tile_dim0 = 64 (K), tile_dim1 = 64 (cols)
//  tensor_dim0_stride = K
__device__ __forceinline__ void tdm_load_b_tile(unsigned int lds_off, unsigned long long gaddr,
                                                int K) {
    u32x4 d0;
    d0[0] = 1u;                                         // count=1 (valid user D#)
    d0[1] = lds_off;                                    // lds_addr
    d0[2] = (unsigned int)gaddr;                        // global_addr[31:0]
    d0[3] = (unsigned int)(gaddr >> 32) | (2u << 30);   // global_addr[56:32] | type=2
    i32x8 d1;
    d1[0] = (int)((1u << 16) | (1u << 20) | (4u << 22) | (3u << 25));
    d1[1] = (int)((unsigned)(K & 0xFFFF) << 16);                         // tensor_dim0 lo16
    d1[2] = (int)(((unsigned)K >> 16) & 0xFFFFu) | (int)(64u << 16);     // tensor_dim0 hi | tensor_dim1 lo
    d1[3] = (int)(64u << 16);                                            // tensor_dim1 hi=0 | tile_dim0=64
    d1[4] = (int)64u;                                                    // tile_dim1=64, tile_dim2=0
    d1[5] = (int)(unsigned)K;                                            // tensor_dim0_stride lo32
    d1[6] = 0;                                                           // stride hi | dim1_stride lo
    d1[7] = 0;
    i32x4 dz4 = {0, 0, 0, 0};
    i32x8 dz8 = {0, 0, 0, 0, 0, 0, 0, 0};
    // 6-arg toolchain variant: (g0, g1, g2, g3, g4, cpol)
    __builtin_amdgcn_tensor_load_to_lds(d0, d1, dz4, dz4, dz8, 0);
}

__global__ __launch_bounds__(256)
void gemm_nt_bf16(const __bf16* __restrict__ A, const __bf16* __restrict__ B,
                  float* __restrict__ C, int M, int N, int K) {
    __shared__ __attribute__((aligned(16))) __bf16 sB[2][BCOLS * BPITCH]; // 2 x 9216 B

    const int lane = threadIdx.x & 31;
    const int wave = threadIdx.x >> 5;
    const int g    = lane >> 4;    // half-wave group (0/1)
    const int lr   = lane & 15;    // row (A) / column (B) within 16
    const int m0   = blockIdx.x * 128 + wave * 16;
    const int n0   = blockIdx.y * 64;

    v8f acc[4] = {};

    const __bf16* arow = A + (size_t)(m0 + lr) * K;
    const unsigned long long gB =
        (unsigned long long)(uintptr_t)(B + (size_t)n0 * K);
    const unsigned int lds0 = (unsigned int)(uintptr_t)&sB[0][0];
    const unsigned int lds1 = (unsigned int)(uintptr_t)&sB[1][0];

    // prologue: DMA stage 0 (wave 0 only; TDM ignores EXEC, gate by wave)
    if (threadIdx.x < 32)
        tdm_load_b_tile(lds0, gB, K);

    for (int k = 0; k < K; k += KSTAGE) {
        const int buf = (k >> 6) & 1;
        if (threadIdx.x < 32)
            __builtin_amdgcn_s_wait_tensorcnt(0);   // stage k landed in LDS
        __syncthreads();                            // publish stage k; retire reads of other buf
        if (threadIdx.x < 32 && (k + KSTAGE) < K)
            tdm_load_b_tile(buf ? lds0 : lds1, gB + 2ull * (k + KSTAGE), K);

        const __bf16* sb = &sB[buf][0];
#pragma unroll
        for (int kk = 0; kk < KSTAGE; kk += 32) {
            // A fragment 16x32 bf16: lane lr = row M.
            // VGPR0-3: K = kk + 8*g + [0..7]; VGPR4-7: K = kk + 16 + 8*g + [0..7]
            union { v16bf v; v8bf h[2]; } a;
            a.h[0] = *(const v8bf*)(arow + k + kk + 8 * g);
            a.h[1] = *(const v8bf*)(arow + k + kk + 16 + 8 * g);
            __builtin_prefetch(arow + k + kk + 256, 0, 3);  // global_prefetch_b8, keep near

#pragma unroll
            for (int t = 0; t < 4; ++t) {
                // B fragment 32x16 (KxN) from LDS: lane lr = column,
                // VGPR0-3: K = kk + 8*g + [0..7]; VGPR4-7: K = kk + 16 + 8*g + [0..7]
                union { v16bf v; v8bf h[2]; } b;
                const __bf16* br = sb + (size_t)(t * 16 + lr) * BPITCH + kk + 8 * g;
                b.h[0] = *(const v8bf*)(br);
                b.h[1] = *(const v8bf*)(br + 16);

                acc[t] = __builtin_amdgcn_wmma_f32_16x16x32_bf16(
                    false, a.v, false, b.v, (short)0, acc[t], false, false);
            }
        }
    }

    // C layout: lanes 0-15: N=lane, VGPR r -> M=r ; lanes 16-31: N=lane-16, M=8+r
#pragma unroll
    for (int t = 0; t < 4; ++t) {
        float* crow = C + (size_t)(m0 + g * 8) * N + (n0 + t * 16 + lr);
#pragma unroll
        for (int r = 0; r < 8; ++r)
            crow[(size_t)r * N] = acc[t][r];
    }
}

// ------------------------- attention score row-dots ----------------------
// s_src[i] = dot(z[i,:], a[0:d]) ; s_dst[i] = dot(z[i,:], a[d:2d])

__global__ __launch_bounds__(256)
void rowdot2(const float* __restrict__ z, const float* __restrict__ a,
             float* __restrict__ s_src, float* __restrict__ s_dst, int d) {
    __shared__ float rA[256];
    __shared__ float rB[256];
    const float* row = z + (size_t)blockIdx.x * d;
    float sa = 0.f, sb = 0.f;
    for (int j = threadIdx.x; j < d; j += 256) {
        float v = row[j];
        sa += v * a[j];
        sb += v * a[j + d];
    }
    rA[threadIdx.x] = sa;
    rB[threadIdx.x] = sb;
    __syncthreads();
    for (int s = 128; s > 0; s >>= 1) {
        if ((int)threadIdx.x < s) {
            rA[threadIdx.x] += rA[threadIdx.x + s];
            rB[threadIdx.x] += rB[threadIdx.x + s];
        }
        __syncthreads();
    }
    if (threadIdx.x == 0) {
        s_src[blockIdx.x] = rA[0];
        s_dst[blockIdx.x] = rB[0];
    }
}

// ------------------------- edge softmax + aggregation --------------------

__device__ __forceinline__ void atomic_max_f32(float* addr, float val) {
    // IEEE ordering trick: int max for >=0, unsigned min for <0 (init = -inf)
    if (!(val < 0.0f))
        __hip_atomic_fetch_max((int*)addr, __float_as_int(val),
                               __ATOMIC_RELAXED, __HIP_MEMORY_SCOPE_AGENT);
    else
        __hip_atomic_fetch_min((unsigned int*)addr, __float_as_uint(val),
                               __ATOMIC_RELAXED, __HIP_MEMORY_SCOPE_AGENT);
}

__device__ __forceinline__ void atomic_add_f32(float* addr, float v) {
    __hip_atomic_fetch_add(addr, v, __ATOMIC_RELAXED, __HIP_MEMORY_SCOPE_AGENT);
}

// Pass 1: per-dst max of e = s_src[src] + s_dst[dst] (only kept dsts)
__global__ void edge_max(const int* __restrict__ src, const int* __restrict__ dst,
                         const float* __restrict__ ssrc, const float* __restrict__ sdst,
                         float* __restrict__ m, int nE, int lo, int hi) {
    int i = blockIdx.x * blockDim.x + threadIdx.x;
    if (i >= nE) return;
    int d = dst[i];
    if (d < lo || d >= hi) return;
    float e = ssrc[src[i]] + sdst[d];
    atomic_max_f32(&m[d], e);
}

// Pass 2: ex = exp(e - m[dst]); denom[dst] += ex; stash ex per edge
__global__ void edge_expsum(const int* __restrict__ src, const int* __restrict__ dst,
                            const float* __restrict__ ssrc, const float* __restrict__ sdst,
                            const float* __restrict__ m, float* __restrict__ denom,
                            float* __restrict__ exbuf, int nE, int lo, int hi) {
    int i = blockIdx.x * blockDim.x + threadIdx.x;
    if (i >= nE) return;
    int d = dst[i];
    if (d < lo || d >= hi) return;
    float e  = ssrc[src[i]] + sdst[d];
    float ex = __expf(e - m[d]);
    exbuf[i] = ex;
    atomic_add_f32(&denom[d], ex);
}

// Pass 3: out[dst - rowoff] += (ex/denom[dst]) * z[src].  One block per edge,
// 256 threads x float4 covers D_OUT=1024 exactly.
__global__ __launch_bounds__(256)
void edge_agg(const int* __restrict__ src, const int* __restrict__ dst,
              const float* __restrict__ exbuf, const float* __restrict__ denom,
              const float* __restrict__ z, float* __restrict__ out,
              int lo, int hi, int rowoff) {
    int e = blockIdx.x;
    int d = dst[e];
    if (d < lo || d >= hi) return;
    float alpha = exbuf[e] / denom[d];
    const float4* zr = (const float4*)(z + (size_t)src[e] * D_OUT);
    float* orow = out + (size_t)(d - rowoff) * D_OUT;
    int j = threadIdx.x;
    float4 v = zr[j];
    atomic_add_f32(&orow[4 * j + 0], alpha * v.x);
    atomic_add_f32(&orow[4 * j + 1], alpha * v.y);
    atomic_add_f32(&orow[4 * j + 2], alpha * v.z);
    atomic_add_f32(&orow[4 * j + 3], alpha * v.w);
}

// ------------------------- host-side orchestration -----------------------

extern "C" void kernel_launch(void* const* d_in, const int* in_sizes, int n_in,
                              void* d_out, int out_size, void* d_ws, size_t ws_size,
                              hipStream_t stream) {
    const float* kn_emb  = (const float*)d_in[0];
    const float* ex_emb  = (const float*)d_in[1];
    const int* dir_src = (const int*)d_in[2];
    const int* dir_dst = (const int*)d_in[3];
    const int* und_src = (const int*)d_in[4];
    const int* und_dst = (const int*)d_in[5];
    const int* kfe_src = (const int*)d_in[6];
    const int* kfe_dst = (const int*)d_in[7];
    const int* efk_src = (const int*)d_in[8];
    const int* efk_dst = (const int*)d_in[9];
    const float* W_dir = (const float*)d_in[10];
    const float* a_dir = (const float*)d_in[11];
    const float* W_und = (const float*)d_in[12];
    const float* a_und = (const float*)d_in[13];
    const float* W_kfe = (const float*)d_in[14];
    const float* a_kfe = (const float*)d_in[15];
    const float* W_efk = (const float*)d_in[16];
    const float* a_efk = (const float*)d_in[17];
    float* out = (float*)d_out;   // [kn_out (1024x1024) | exer_out (20480x1024)]

    // workspace carve-up
    char* ws = (char*)d_ws;
    size_t off = 0;
    auto take = [&](size_t bytes) -> char* {
        char* p = ws + off;
        off += (bytes + 255) & ~(size_t)255;
        return p;
    };
    __bf16* hb     = (__bf16*)take((size_t)NB * D_IN * sizeof(__bf16)); // [exer ; kn]
    __bf16* wb_dir = (__bf16*)take((size_t)D_OUT * D_IN * sizeof(__bf16));
    __bf16* wb_und = (__bf16*)take((size_t)D_OUT * D_IN * sizeof(__bf16));
    __bf16* wb_kfe = (__bf16*)take((size_t)D_OUT * D_IN * sizeof(__bf16));
    __bf16* wb_efk = (__bf16*)take((size_t)D_OUT * D_IN * sizeof(__bf16));
    float* z_dir = (float*)take((size_t)N_K * D_OUT * 4);
    float* z_und = (float*)take((size_t)N_K * D_OUT * 4);
    float* z_kfe = (float*)take((size_t)NB  * D_OUT * 4);
    float* z_efk = (float*)take((size_t)NB  * D_OUT * 4);
    float* ss_dir = (float*)take(N_K * 4); float* sd_dir = (float*)take(N_K * 4);
    float* ss_und = (float*)take(N_K * 4); float* sd_und = (float*)take(N_K * 4);
    float* ss_kfe = (float*)take((size_t)NB * 4); float* sd_kfe = (float*)take((size_t)NB * 4);
    float* ss_efk = (float*)take((size_t)NB * 4); float* sd_efk = (float*)take((size_t)NB * 4);
    float* m_dir = (float*)take(N_K * 4); float* dn_dir = (float*)take(N_K * 4);
    float* m_und = (float*)take(N_K * 4); float* dn_und = (float*)take(N_K * 4);
    float* m_kfe = (float*)take((size_t)NB * 4); float* dn_kfe = (float*)take((size_t)NB * 4);
    float* m_efk = (float*)take((size_t)NB * 4); float* dn_efk = (float*)take((size_t)NB * 4);
    float* ex_dir = (float*)take((size_t)E_K * 4);
    float* ex_und = (float*)take((size_t)E_K * 4);
    float* ex_kfe = (float*)take((size_t)E_B * 4);
    float* ex_efk = (float*)take((size_t)E_B * 4);

    const __bf16* hb_kn = hb + (size_t)N_E * D_IN;   // kn rows inside concat
    const float NEG_INF = -__builtin_huge_valf();

    // --- fp32 -> bf16 conversions ---
    cvt_f32_bf16<<<2048, 256, 0, stream>>>(ex_emb, hb,             (size_t)N_E * D_IN);
    cvt_f32_bf16<<<1024, 256, 0, stream>>>(kn_emb, (__bf16*)hb_kn, (size_t)N_K * D_IN);
    cvt_f32_bf16<<<1024, 256, 0, stream>>>(W_dir, wb_dir, (size_t)D_OUT * D_IN);
    cvt_f32_bf16<<<1024, 256, 0, stream>>>(W_und, wb_und, (size_t)D_OUT * D_IN);
    cvt_f32_bf16<<<1024, 256, 0, stream>>>(W_kfe, wb_kfe, (size_t)D_OUT * D_IN);
    cvt_f32_bf16<<<1024, 256, 0, stream>>>(W_efk, wb_efk, (size_t)D_OUT * D_IN);

    // --- z = h @ W^T via WMMA bf16 + TDM-staged B ---
    {
        dim3 blk(256);
        dim3 g_small(N_K / 128, D_OUT / 64);   // 8 x 16
        dim3 g_big(NB / 128, D_OUT / 64);      // 168 x 16
        gemm_nt_bf16<<<g_small, blk, 0, stream>>>(hb_kn, wb_dir, z_dir, N_K, D_OUT, D_IN);
        gemm_nt_bf16<<<g_small, blk, 0, stream>>>(hb_kn, wb_und, z_und, N_K, D_OUT, D_IN);
        gemm_nt_bf16<<<g_big,   blk, 0, stream>>>(hb,    wb_kfe, z_kfe, NB,  D_OUT, D_IN);
        gemm_nt_bf16<<<g_big,   blk, 0, stream>>>(hb,    wb_efk, z_efk, NB,  D_OUT, D_IN);
    }

    // --- attention score row dots ---
    rowdot2<<<N_K, 256, 0, stream>>>(z_dir, a_dir, ss_dir, sd_dir, D_OUT);
    rowdot2<<<N_K, 256, 0, stream>>>(z_und, a_und, ss_und, sd_und, D_OUT);
    rowdot2<<<NB,  256, 0, stream>>>(z_kfe, a_kfe, ss_kfe, sd_kfe, D_OUT);
    rowdot2<<<NB,  256, 0, stream>>>(z_efk, a_efk, ss_efk, sd_efk, D_OUT);

    // --- init reductions + output ---
    fill_f32<<<64, 256, 0, stream>>>(m_dir, NEG_INF, N_K);
    fill_f32<<<64, 256, 0, stream>>>(m_und, NEG_INF, N_K);
    fill_f32<<<128, 256, 0, stream>>>(m_kfe, NEG_INF, (size_t)NB);
    fill_f32<<<128, 256, 0, stream>>>(m_efk, NEG_INF, (size_t)NB);
    fill_f32<<<64, 256, 0, stream>>>(dn_dir, 0.f, N_K);
    fill_f32<<<64, 256, 0, stream>>>(dn_und, 0.f, N_K);
    fill_f32<<<128, 256, 0, stream>>>(dn_kfe, 0.f, (size_t)NB);
    fill_f32<<<128, 256, 0, stream>>>(dn_efk, 0.f, (size_t)NB);
    fill_f32<<<4096, 256, 0, stream>>>(out, 0.f, (size_t)NB * D_OUT);

    // --- segment max (only dsts that survive to the output) ---
    edge_max<<<E_K / 256, 256, 0, stream>>>(dir_src, dir_dst, ss_dir, sd_dir, m_dir, E_K, 0, N_K);
    edge_max<<<E_K / 256, 256, 0, stream>>>(und_src, und_dst, ss_und, sd_und, m_und, E_K, 0, N_K);
    edge_max<<<E_B / 256, 256, 0, stream>>>(kfe_src, kfe_dst, ss_kfe, sd_kfe, m_kfe, E_B, N_E, NB);
    edge_max<<<E_B / 256, 256, 0, stream>>>(efk_src, efk_dst, ss_efk, sd_efk, m_efk, E_B, 0, N_E);

    // --- exp + denom ---
    edge_expsum<<<E_K / 256, 256, 0, stream>>>(dir_src, dir_dst, ss_dir, sd_dir, m_dir, dn_dir, ex_dir, E_K, 0, N_K);
    edge_expsum<<<E_K / 256, 256, 0, stream>>>(und_src, und_dst, ss_und, sd_und, m_und, dn_und, ex_und, E_K, 0, N_K);
    edge_expsum<<<E_B / 256, 256, 0, stream>>>(kfe_src, kfe_dst, ss_kfe, sd_kfe, m_kfe, dn_kfe, ex_kfe, E_B, N_E, NB);
    edge_expsum<<<E_B / 256, 256, 0, stream>>>(efk_src, efk_dst, ss_efk, sd_efk, m_efk, dn_efk, ex_efk, E_B, 0, N_E);

    // --- weighted scatter-aggregate, fused into d_out ---
    float* out_kn = out;                              // rows = kn ids
    float* out_ex = out + (size_t)N_K * D_OUT;        // rows = exer ids
    edge_agg<<<E_K, 256, 0, stream>>>(dir_src, dir_dst, ex_dir, dn_dir, z_dir, out_kn, 0, N_K, 0);
    edge_agg<<<E_K, 256, 0, stream>>>(und_src, und_dst, ex_und, dn_und, z_und, out_kn, 0, N_K, 0);
    edge_agg<<<E_B, 256, 0, stream>>>(kfe_src, kfe_dst, ex_kfe, dn_kfe, z_kfe, out_kn, N_E, NB, N_E);
    edge_agg<<<E_B, 256, 0, stream>>>(efk_src, efk_dst, ex_efk, dn_efk, z_efk, out_ex, 0, N_E, 0);

    (void)in_sizes; (void)n_in; (void)out_size; (void)ws_size;
}
